// GTLayer_25056839204915
// MI455X (gfx1250) — matrix operand order repaired and checked by reading, not compiled, not run on hardware
//
#include <hip/hip_runtime.h>
#include <hip/hip_bf16.h>

#define N_NODES 50000
#define N_EDGES 800000
#define LATDIM  128
#define HEAD    4

typedef __attribute__((ext_vector_type(16))) __bf16 v16bf;
typedef __attribute__((ext_vector_type(8)))  float  v8f;

__global__ void zero_f32_kernel(float* __restrict__ p, int n) {
    int i = blockIdx.x * blockDim.x + threadIdx.x;
    if (i < n) p[i] = 0.0f;
}

// Per-node QKV projection: Y = embeds @ W for W in {qW,kW,vW}.
// One wave per 16x16 output tile; 8 waves/block = one 16-row tile x all 8 col tiles.
// gridDim.x = (N_NODES/16) * 3
__global__ void qkv_gemm_kernel(const float* __restrict__ embeds,
                                const float* __restrict__ qW,
                                const float* __restrict__ kW,
                                const float* __restrict__ vW,
                                float* __restrict__ Q,
                                float* __restrict__ K,
                                float* __restrict__ V) {
    const int mat     = blockIdx.x % 3;
    const int rowTile = blockIdx.x / 3;
    const float* __restrict__ W = (mat == 0) ? qW : (mat == 1) ? kW : vW;
    float* __restrict__ Y       = (mat == 0) ? Q  : (mat == 1) ? K  : V;

    const int wave = threadIdx.x >> 5;     // 0..7 -> col tile
    const int lane = threadIdx.x & 31;
    const int half = lane >> 4;            // 0: lanes 0-15, 1: lanes 16-31
    const int m    = lane & 15;
    const int row0 = rowTile * 16;
    const int col0 = wave * 16;

    v8f c = {};
    #pragma unroll
    for (int kc = 0; kc < 4; ++kc) {
        const int kbase = kc * 32;

        // ---- A fragment: 16x32 bf16, M = m (contiguous k order, same as B) ----
        // lane<16 : idx 0..15 -> K=kbase+0..15
        // lane>=16: idx 0..15 -> K=kbase+16..31
        const float* arow = embeds + (size_t)(row0 + m) * LATDIM + kbase + half * 16;
        float4 a0 = *(const float4*)(arow +  0);
        float4 a1 = *(const float4*)(arow +  4);
        float4 a2 = *(const float4*)(arow +  8);
        float4 a3 = *(const float4*)(arow + 12);
        v16bf a;
        a[0]  = (__bf16)a0.x; a[1]  = (__bf16)a0.y;
        a[2]  = (__bf16)a0.z; a[3]  = (__bf16)a0.w;
        a[4]  = (__bf16)a1.x; a[5]  = (__bf16)a1.y;
        a[6]  = (__bf16)a1.z; a[7]  = (__bf16)a1.w;
        a[8]  = (__bf16)a2.x; a[9]  = (__bf16)a2.y;
        a[10] = (__bf16)a2.z; a[11] = (__bf16)a2.w;
        a[12] = (__bf16)a3.x; a[13] = (__bf16)a3.y;
        a[14] = (__bf16)a3.z; a[15] = (__bf16)a3.w;

        // ---- B fragment: 32x16 bf16, N = m, K = kbase + half*16 + i ----
        const float* wcol = W + (size_t)(kbase + half * 16) * LATDIM + col0 + m;
        float wv[16];
        #pragma unroll
        for (int i = 0; i < 16; ++i) wv[i] = wcol[(size_t)i * LATDIM];
        v16bf b;
        #pragma unroll
        for (int i = 0; i < 16; ++i) b[i] = (__bf16)wv[i];

        c = __builtin_amdgcn_wmma_f32_16x16x32_bf16(
                /*neg_a=*/false, a, /*neg_b=*/false, b,
                /*c_mod=*/(short)0, c, /*reuse_a=*/false, /*reuse_b=*/false);
    }

    // ---- D store: c[g] -> Y[row0 + g + 8*half, col0 + m] ----
    #pragma unroll
    for (int g = 0; g < 8; ++g)
        Y[(size_t)(row0 + g + half * 8) * LATDIM + col0 + m] = c[g];
}

// Pass A: one wave per edge. lane*4 covers all 128 dims; head = lane/8.
// group-of-8 shuffle reduction gives per-head q.k; clip, exp, scatter-add norm.
__global__ void edge_att_kernel(const float* __restrict__ Q,
                                const float* __restrict__ Kp,
                                const int* __restrict__ rows,
                                const int* __restrict__ cols,
                                float* __restrict__ expAtt,
                                float* __restrict__ attNorm) {
    const int e    = (blockIdx.x * blockDim.x + threadIdx.x) >> 5;
    const int lane = threadIdx.x & 31;
    if (e >= N_EDGES) return;
    const int r = rows[e];
    const int c = cols[e];

    float4 q = *(const float4*)(Q  + (size_t)r * LATDIM + lane * 4);
    float4 k = *(const float4*)(Kp + (size_t)c * LATDIM + lane * 4);
    float p = q.x * k.x + q.y * k.y + q.z * k.z + q.w * k.w;
    p += __shfl_xor(p, 1, 32);
    p += __shfl_xor(p, 2, 32);
    p += __shfl_xor(p, 4, 32);   // full head-dot in every lane of each 8-lane group

    if ((lane & 7) == 0) {
        const int h = lane >> 3;
        p = fminf(fmaxf(p, -10.0f), 10.0f);
        const float ex = expf(p);
        expAtt[(size_t)e * HEAD + h] = ex;
        atomicAdd(attNorm + (size_t)r * HEAD + h, ex);
    }
}

// Pass B: one wave per edge. Scale gathered V by softmax weight, atomic scatter to out.
__global__ void edge_agg_kernel(const float* __restrict__ V,
                                const int* __restrict__ rows,
                                const int* __restrict__ cols,
                                const float* __restrict__ expAtt,
                                const float* __restrict__ attNorm,
                                float* __restrict__ out) {
    const int e    = (blockIdx.x * blockDim.x + threadIdx.x) >> 5;
    const int lane = threadIdx.x & 31;
    if (e >= N_EDGES) return;
    const int r = rows[e];
    const int c = cols[e];
    const int h = lane >> 3;

    const float a = expAtt[(size_t)e * HEAD + h] /
                    (attNorm[(size_t)r * HEAD + h] + 1e-8f);
    float4 v = *(const float4*)(V + (size_t)c * LATDIM + lane * 4);
    float* o = out + (size_t)r * LATDIM + lane * 4;
    atomicAdd(o + 0, a * v.x);
    atomicAdd(o + 1, a * v.y);
    atomicAdd(o + 2, a * v.z);
    atomicAdd(o + 3, a * v.w);
}

extern "C" void kernel_launch(void* const* d_in, const int* in_sizes, int n_in,
                              void* d_out, int out_size, void* d_ws, size_t ws_size,
                              hipStream_t stream) {
    const float* embeds = (const float*)d_in[0];
    const float* qW     = (const float*)d_in[1];
    const float* kW     = (const float*)d_in[2];
    const float* vW     = (const float*)d_in[3];
    const int*   rows   = (const int*)d_in[4];
    const int*   cols   = (const int*)d_in[5];
    float*       out    = (float*)d_out;

    // Workspace layout (floats): Q | K | V | attNorm | expAtt  (~90.4 MB)
    float* Q       = (float*)d_ws;
    float* K       = Q + (size_t)N_NODES * LATDIM;
    float* V       = K + (size_t)N_NODES * LATDIM;
    float* attNorm = V + (size_t)N_NODES * LATDIM;
    float* expAtt  = attNorm + (size_t)N_NODES * HEAD;

    // 1) zero accumulators (out is poisoned; attNorm in ws)
    {
        int n = N_NODES * LATDIM;
        zero_f32_kernel<<<(n + 255) / 256, 256, 0, stream>>>(out, n);
        int m = N_NODES * HEAD;
        zero_f32_kernel<<<(m + 255) / 256, 256, 0, stream>>>(attNorm, m);
    }

    // 2) per-node QKV projection via WMMA bf16 (50000 rows = 3125 tiles of 16)
    {
        dim3 grid((N_NODES / 16) * 3);
        qkv_gemm_kernel<<<grid, 256, 0, stream>>>(embeds, qW, kW, vW, Q, K, V);
    }

    // 3) edge attention logits + segment-softmax denominator (wave per edge)
    {
        long long threads = (long long)N_EDGES * 32;
        int blocks = (int)((threads + 255) / 256);
        edge_att_kernel<<<blocks, 256, 0, stream>>>(Q, K, rows, cols, expAtt, attNorm);
    }

    // 4) normalized weighted scatter of V into out (wave per edge)
    {
        long long threads = (long long)N_EDGES * 32;
        int blocks = (int)((threads + 255) / 256);
        edge_agg_kernel<<<blocks, 256, 0, stream>>>(V, rows, cols, expAtt, attNorm, out);
    }
}
